// TaskConditionedAttention_47528108097692
// MI455X (gfx1250) — matrix operand order repaired and checked by reading, not compiled, not run on hardware
//
#include <hip/hip_runtime.h>
#include <cstdint>

#define B_  4
#define S_  2048
#define D_  1024
#define H_  16
#define HD_ 64
#define TE_ 64

typedef __attribute__((ext_vector_type(16))) __bf16 v16bf;
typedef __attribute__((ext_vector_type(8)))  __bf16 v8bf;
typedef __attribute__((ext_vector_type(8)))  float  v8f;
typedef __attribute__((ext_vector_type(4)))  unsigned int v4u;
typedef __attribute__((ext_vector_type(4)))  int v4i;
typedef __attribute__((ext_vector_type(8)))  int v8i;

__device__ inline v8f zero_v8f() {
  v8f z;
#pragma unroll
  for (int i = 0; i < 8; ++i) z[i] = 0.0f;
  return z;
}

// Async global->LDS copy of 16 bytes (CDNA5 ASYNCcnt path, no VGPR roundtrip).
__device__ inline void async_b128(const __bf16* gptr, const __bf16* lptr) {
  const uint32_t lds = (uint32_t)(uintptr_t)lptr;  // low 32b = LDS offset
  asm volatile("global_load_async_to_lds_b128 %0, %1, off"
               :: "v"(lds), "v"(gptr) : "memory");
}
__device__ inline void wait_async0() {
  asm volatile("s_wait_asynccnt 0x0" ::: "memory");
}

// Tensor Data Mover: one descriptor moves a 64x64 bf16 tile (row stride
// `stride` elements in global) into contiguous LDS. D# layout per ISA §8.3/8.4.
__device__ inline void tdm_load_tile_64x64_bf16(const __bf16* gsrc,
                                                const __bf16* ldst,
                                                int stride_elems) {
  union { v4u v; uint32_t u[4]; } g0;
  const uint64_t ga = (uint64_t)(uintptr_t)gsrc;
  g0.u[0] = 1u;                          // count=1, user mode, no gather
  g0.u[1] = (uint32_t)(uintptr_t)ldst;   // lds_addr (bytes)
  g0.u[2] = (uint32_t)ga;                // global_addr[31:0]
  g0.u[3] = (uint32_t)((ga >> 32) & 0x01FFFFFFu) | (2u << 30);  // addr[56:32], type=2

  union { v8i v; uint32_t u[8]; } g1;
  g1.u[0] = 1u << 16;                    // wg_mask=0, data_size=1 (2 bytes)
  g1.u[1] = 64u << 16;                   // tensor_dim0[15:0] = 64 at bits 63:48
  g1.u[2] = 64u << 16;                   // tensor_dim1[15:0] = 64 at bits 95:80
  g1.u[3] = 64u << 16;                   // tile_dim0 = 64 at bits 127:112
  g1.u[4] = 64u;                         // tile_dim1 = 64, tile_dim2 = 0
  g1.u[5] = (uint32_t)stride_elems;      // tensor_dim0_stride[31:0]
  g1.u[6] = 0u;                          // stride[47:32]=0, dim1_stride[15:0]=0
  g1.u[7] = 0u;

  const v4i z4 = {0, 0, 0, 0};
#if defined(__clang_major__) && (__clang_major__ >= 23)
  const v8i z8 = {0, 0, 0, 0, 0, 0, 0, 0};
  __builtin_amdgcn_tensor_load_to_lds(g0.v, g1.v, z4, z4, z8, 0);
#else
  __builtin_amdgcn_tensor_load_to_lds(g0.v, g1.v, z4, z4, 0);
#endif
}

// A-fragment (16x32 bf16). Lane layout (ISA 7.12.2): lanes 0-15 row=l16,
// cols: e0..7 -> K=e+half*8, e8..15 -> K=16+half*8+(e-8). Two 16B LDS loads.
__device__ inline v16bf load_a_frag(const __bf16* rowbase, int half) {
  v8bf lo = *(const v8bf*)(rowbase + half * 8);
  v8bf hi = *(const v8bf*)(rowbase + 16 + half * 8);
  v16bf r;
#pragma unroll
  for (int i = 0; i < 8; ++i) { r[i] = lo[i]; r[i + 8] = hi[i]; }
  return r;
}

// B-fragment (32x16 bf16). Source tile N-major/K-contiguous:
// lane holds col N=l16, e0..7 -> K=e+half*8, e8..15 -> K=16+half*8+(e-8).
__device__ inline v16bf load_b_frag(const __bf16* rowbase, int half) {
  v8bf lo = *(const v8bf*)(rowbase + half * 8);
  v8bf hi = *(const v8bf*)(rowbase + 16 + half * 8);
  v16bf r;
#pragma unroll
  for (int i = 0; i < 8; ++i) { r[i] = lo[i]; r[i + 8] = hi[i]; }
  return r;
}

// ---------------------------------------------------------------- converts
__global__ void cvt_bf16_kernel(const float* __restrict__ in,
                                __bf16* __restrict__ out, int n) {
  int i = blockIdx.x * blockDim.x + threadIdx.x;
  if (i < n) out[i] = (__bf16)in[i];
}

// W (D x D, row-major k,n) -> Wt (N x K, row-major), bf16. 32x32 LDS tiles.
__global__ void cvt_bf16_transpose_kernel(const float* __restrict__ W,
                                          __bf16* __restrict__ Wt) {
  __shared__ float tile[32][33];
  const int kt = blockIdx.y * 32, nt = blockIdx.x * 32;
#pragma unroll
  for (int i = threadIdx.y; i < 32; i += 8)
    tile[i][threadIdx.x] = W[(size_t)(kt + i) * D_ + nt + threadIdx.x];
  __syncthreads();
#pragma unroll
  for (int i = threadIdx.y; i < 32; i += 8)
    Wt[(size_t)(nt + i) * D_ + kt + threadIdx.x] = (__bf16)tile[threadIdx.x][i];
}

// task_bias[b][h] = task_emb[b] . Wt[:,h] + bt[h]    (B*H = 64 values)
__global__ void task_bias_kernel(const float* __restrict__ te,
                                 const float* __restrict__ Wt,
                                 const float* __restrict__ bt,
                                 float* __restrict__ out) {
  const int i = threadIdx.x;  // 0..63
  const int b = i / H_, h = i % H_;
  float s = bt[h];
#pragma unroll 8
  for (int t = 0; t < TE_; ++t) s += te[b * TE_ + t] * Wt[t * H_ + h];
  out[i] = s;
}

// ---------------------------------------------------------------- GEMM
// Out[M,N] = A[M,K] @ Bt[N,K]^T + bias.  Block tile 128x128, K-step 32,
// 256 threads = 8 waves (each 32x64 sub-tile = 2x4 WMMA frags).
// Double-buffered LDS fed by global_load_async_to_lds_b128: the next K-tile
// streams into buf^1 while WMMAs consume buf.
template <bool F32OUT>
__global__ __launch_bounds__(256) void gemm_bf16_kernel(
    const __bf16* __restrict__ A, const __bf16* __restrict__ Bt,
    const float* __restrict__ bias, void* __restrict__ Out,
    int M, int N, int K) {
  __shared__ __align__(16) __bf16 sA[2][128 * 32];
  __shared__ __align__(16) __bf16 sB[2][128 * 32];

  const int tid  = threadIdx.x;
  const int lane = tid & 31, wave = tid >> 5;
  const int half = lane >> 4, l16 = lane & 15;
  const int wm = wave & 3;   // 32-row slab
  const int wn = wave >> 2;  // 64-col slab
  const int rowBase = blockIdx.y * 128;
  const int colBase = blockIdx.x * 128;

  v8f acc[2][4];
#pragma unroll
  for (int mi = 0; mi < 2; ++mi)
#pragma unroll
    for (int ni = 0; ni < 4; ++ni) acc[mi][ni] = zero_v8f();

  // Tile = 128x32 bf16 = 512 16B-chunks; 256 threads -> 2 chunks per array.
  auto issue_tile = [&](int k0, int buf) {
#pragma unroll
    for (int c = tid; c < 512; c += 256) {
      const int r = c >> 2, co = (c & 3) * 8;
      async_b128(A  + (size_t)(rowBase + r) * K + k0 + co, &sA[buf][r * 32 + co]);
      async_b128(Bt + (size_t)(colBase + r) * K + k0 + co, &sB[buf][r * 32 + co]);
    }
  };

  issue_tile(0, 0);
  wait_async0();
  __syncthreads();

  for (int k0 = 0; k0 < K; k0 += 32) {
    const int buf = (k0 >> 5) & 1;
    if (k0 + 32 < K) issue_tile(k0 + 32, buf ^ 1);  // overlap with compute

    v16bf af[2], bf[4];
#pragma unroll
    for (int mi = 0; mi < 2; ++mi)
      af[mi] = load_a_frag(&sA[buf][(wm * 32 + mi * 16 + l16) * 32], half);
#pragma unroll
    for (int ni = 0; ni < 4; ++ni)
      bf[ni] = load_b_frag(&sB[buf][(wn * 64 + ni * 16 + l16) * 32], half);
#pragma unroll
    for (int mi = 0; mi < 2; ++mi)
#pragma unroll
      for (int ni = 0; ni < 4; ++ni)
        acc[mi][ni] = __builtin_amdgcn_wmma_f32_16x16x32_bf16(
            false, af[mi], false, bf[ni], (short)0, acc[mi][ni], false, false);

    wait_async0();   // next tile landed in LDS
    __syncthreads();
  }

  // C/D layout: lane col N=l16, VGPR j -> row j + half*8
#pragma unroll
  for (int mi = 0; mi < 2; ++mi) {
#pragma unroll
    for (int ni = 0; ni < 4; ++ni) {
      const int col = colBase + wn * 64 + ni * 16 + l16;
      const float bv = bias[col];
#pragma unroll
      for (int j = 0; j < 8; ++j) {
        const int row = rowBase + wm * 32 + mi * 16 + half * 8 + j;
        const float v = acc[mi][ni][j] + bv;
        if (F32OUT) ((float*)Out)[(size_t)row * N + col] = v;
        else        ((__bf16*)Out)[(size_t)row * N + col] = (__bf16)v;
      }
    }
  }
}

// ---------------------------------------------------------------- attention
// One block = (b,h, 64 q-rows). 4 waves, each wave owns 16 q-rows; online
// softmax over all S keys in 64-key tiles. Q streams via async-to-LDS, the
// K tile rides the Tensor Data Mover (one D# per 64-key step, TENSORcnt),
// and V is transposed manually into [hd][key] while the TDM copy is in flight.
__global__ __launch_bounds__(128) void attn_kernel(
    const __bf16* __restrict__ Qb, const __bf16* __restrict__ Kb,
    const __bf16* __restrict__ Vb, const int* __restrict__ mask,
    const float* __restrict__ tbias, __bf16* __restrict__ Ob) {
  __shared__ __align__(16) __bf16 sQ[64 * 64];
  __shared__ __align__(16) __bf16 sK[64 * 64];     // [key][hd] (B-frag ready)
  __shared__ __align__(16) __bf16 sV[64 * 64];     // [hd][key] (transposed)
  __shared__ __align__(16) __bf16 sP[4][16 * 64];  // per-wave P tile

  const int tid  = threadIdx.x;
  const int lane = tid & 31, wave = tid >> 5;
  const int half = lane >> 4, l16 = lane & 15;

  const int nqt = S_ / 64;
  const int qt = blockIdx.x % nqt;
  const int bh = blockIdx.x / nqt;
  const int b = bh / H_, h = bh % H_;

  const float tb = tbias[b * H_ + h];
  const float scale = 0.125f;  // HD^-0.5

  // Q tile: 64x64 bf16 = 512 16B-chunks, 128 threads -> 4 chunks each.
#pragma unroll
  for (int c = tid; c < 512; c += 128) {
    const int r = c >> 3, co = (c & 7) * 8;
    async_b128(Qb + (size_t)(b * S_ + qt * 64 + r) * D_ + h * HD_ + co,
               &sQ[r * 64 + co]);
  }
  wait_async0();
  __syncthreads();

  v16bf qf[2];
#pragma unroll
  for (int kf = 0; kf < 2; ++kf)
    qf[kf] = load_a_frag(&sQ[(wave * 16 + l16) * 64 + kf * 32], half);

  float m[8], l[8];
  v8f o[4];
#pragma unroll
  for (int j = 0; j < 8; ++j) { m[j] = -1e30f; l[j] = 0.0f; }
#pragma unroll
  for (int ni = 0; ni < 4; ++ni) o[ni] = zero_v8f();

  const int ldr = tid >> 1, seg = tid & 1;  // V loader: 64 rows x 32-elt halves

  for (int kb = 0; kb < S_; kb += 64) {
    // K tile via the Tensor Data Mover: one 64x64 2D descriptor, issued by
    // wave 0 (TDM ignores EXEC; one issue per workgroup is enough).
    if (wave == 0)
      tdm_load_tile_64x64_bf16(Kb + (size_t)(b * S_ + kb) * D_ + h * HD_,
                               sK, D_);
    {  // V tile: manual transpose into [hd][key] (overlaps the TDM copy)
      const __bf16* g = Vb + (size_t)(b * S_ + kb + ldr) * D_ + h * HD_ + seg * 32;
      v8bf r0 = ((const v8bf*)g)[0], r1 = ((const v8bf*)g)[1];
      v8bf r2 = ((const v8bf*)g)[2], r3 = ((const v8bf*)g)[3];
#pragma unroll
      for (int i = 0; i < 8; ++i) {
        sV[(seg * 32 + i)      * 64 + ldr] = r0[i];
        sV[(seg * 32 + 8 + i)  * 64 + ldr] = r1[i];
        sV[(seg * 32 + 16 + i) * 64 + ldr] = r2[i];
        sV[(seg * 32 + 24 + i) * 64 + ldr] = r3[i];
      }
    }
    if (wave == 0) __builtin_amdgcn_s_wait_tensorcnt(0);
    __syncthreads();

    // S = Q @ K^T  (HD=64 -> two K=32 WMMAs per 16x16 score tile)
    float sc[4][8];
#pragma unroll
    for (int ni = 0; ni < 4; ++ni) {
      v8f s = zero_v8f();
#pragma unroll
      for (int kf = 0; kf < 2; ++kf) {
        v16bf kfr = load_b_frag(&sK[(ni * 16 + l16) * 64 + kf * 32], half);
        s = __builtin_amdgcn_wmma_f32_16x16x32_bf16(
            false, qf[kf], false, kfr, (short)0, s, false, false);
      }
      const int key = kb + ni * 16 + l16;  // this lane's key column
      const int mk = mask[b * S_ + key];
#pragma unroll
      for (int j = 0; j < 8; ++j)
        sc[ni][j] = (mk == 0) ? -1e30f : (s[j] * scale + tb);
    }

    // online softmax: rows live in VGPR index j (+half*8); reduce over the
    // 16 lanes of each half (xor 1/2/4/8 stays inside the half, wave32).
    float rm[8], fac[8], rs[8];
#pragma unroll
    for (int j = 0; j < 8; ++j) {
      float v = fmaxf(fmaxf(sc[0][j], sc[1][j]), fmaxf(sc[2][j], sc[3][j]));
#pragma unroll
      for (int off = 1; off < 16; off <<= 1)
        v = fmaxf(v, __shfl_xor(v, off, 32));
      rm[j] = v;
    }
#pragma unroll
    for (int j = 0; j < 8; ++j) {
      const float mn = fmaxf(m[j], rm[j]);
      fac[j] = __expf(m[j] - mn);
      m[j] = mn;
      float sum = 0.0f;
#pragma unroll
      for (int ni = 0; ni < 4; ++ni) {
        const float p = __expf(sc[ni][j] - mn);
        sc[ni][j] = p;
        sum += p;
      }
      rs[j] = sum;
    }
#pragma unroll
    for (int j = 0; j < 8; ++j) {
#pragma unroll
      for (int off = 1; off < 16; off <<= 1)
        rs[j] += __shfl_xor(rs[j], off, 32);
      l[j] = l[j] * fac[j] + rs[j];
    }
#pragma unroll
    for (int ni = 0; ni < 4; ++ni)
#pragma unroll
      for (int j = 0; j < 8; ++j) o[ni][j] *= fac[j];

    // P (C-layout) -> per-wave LDS -> A-layout frags (wave-local; LDS ops
    // from one wave are in-order, no cross-wave sharing of sP[wave]).
#pragma unroll
    for (int ni = 0; ni < 4; ++ni)
#pragma unroll
      for (int j = 0; j < 8; ++j)
        sP[wave][(half * 8 + j) * 64 + ni * 16 + l16] = (__bf16)sc[ni][j];

#pragma unroll
    for (int kf = 0; kf < 2; ++kf) {
      v16bf pf = load_a_frag(&sP[wave][l16 * 64 + kf * 32], half);
#pragma unroll
      for (int ni = 0; ni < 4; ++ni) {
        v16bf vf = load_b_frag(&sV[(ni * 16 + l16) * 64 + kf * 32], half);
        o[ni] = __builtin_amdgcn_wmma_f32_16x16x32_bf16(
            false, pf, false, vf, (short)0, o[ni], false, false);
      }
    }
    __syncthreads();
  }

  float inv[8];
#pragma unroll
  for (int j = 0; j < 8; ++j) inv[j] = (l[j] > 0.0f) ? 1.0f / l[j] : 0.0f;
#pragma unroll
  for (int ni = 0; ni < 4; ++ni) {
    const int col = h * HD_ + ni * 16 + l16;
#pragma unroll
    for (int j = 0; j < 8; ++j) {
      const int row = qt * 64 + wave * 16 + half * 8 + j;
      Ob[(size_t)(b * S_ + row) * D_ + col] = (__bf16)(o[ni][j] * inv[j]);
    }
  }
}

// ---------------------------------------------------------------- launch
extern "C" void kernel_launch(void* const* d_in, const int* in_sizes, int n_in,
                              void* d_out, int out_size, void* d_ws, size_t ws_size,
                              hipStream_t stream) {
  (void)in_sizes; (void)n_in; (void)out_size; (void)ws_size;
  const float* x    = (const float*)d_in[0];
  const float* te   = (const float*)d_in[1];
  const int*   mask = (const int*)d_in[2];
  const float* Wq   = (const float*)d_in[3];
  const float* bq   = (const float*)d_in[4];
  const float* Wk   = (const float*)d_in[5];
  const float* bk   = (const float*)d_in[6];
  const float* Wv   = (const float*)d_in[7];
  const float* bv   = (const float*)d_in[8];
  const float* Wo   = (const float*)d_in[9];
  const float* bo   = (const float*)d_in[10];
  const float* Wt   = (const float*)d_in[11];
  const float* bt   = (const float*)d_in[12];
  float* out = (float*)d_out;

  const size_t BSD = (size_t)B_ * S_ * D_;   // 8 Mi elems
  const size_t DD  = (size_t)D_ * D_;        // 1 Mi elems
  __bf16* xb  = (__bf16*)d_ws;
  __bf16* wqT = xb  + BSD;
  __bf16* wkT = wqT + DD;
  __bf16* wvT = wkT + DD;
  __bf16* woT = wvT + DD;
  __bf16* qb  = woT + DD;
  __bf16* kb  = qb  + BSD;
  __bf16* vb  = kb  + BSD;
  __bf16* ob  = vb  + BSD;
  float*  tb  = (float*)(ob + BSD);

  cvt_bf16_kernel<<<(int)(BSD / 256), 256, 0, stream>>>(x, xb, (int)BSD);
  dim3 tgrid(D_ / 32, D_ / 32), tblk(32, 8);
  cvt_bf16_transpose_kernel<<<tgrid, tblk, 0, stream>>>(Wq, wqT);
  cvt_bf16_transpose_kernel<<<tgrid, tblk, 0, stream>>>(Wk, wkT);
  cvt_bf16_transpose_kernel<<<tgrid, tblk, 0, stream>>>(Wv, wvT);
  cvt_bf16_transpose_kernel<<<tgrid, tblk, 0, stream>>>(Wo, woT);
  task_bias_kernel<<<1, B_ * H_, 0, stream>>>(te, Wt, bt, tb);

  dim3 ggrid(D_ / 128, (B_ * S_) / 128);  // (8, 64)
  gemm_bf16_kernel<false><<<ggrid, 256, 0, stream>>>(xb, wqT, bq, (void*)qb, B_ * S_, D_, D_);
  gemm_bf16_kernel<false><<<ggrid, 256, 0, stream>>>(xb, wkT, bk, (void*)kb, B_ * S_, D_, D_);
  gemm_bf16_kernel<false><<<ggrid, 256, 0, stream>>>(xb, wvT, bv, (void*)vb, B_ * S_, D_, D_);

  attn_kernel<<<B_ * H_ * (S_ / 64), 128, 0, stream>>>(qb, kb, vb, mask, tb, ob);

  gemm_bf16_kernel<true><<<ggrid, 256, 0, stream>>>(ob, woT, bo, (void*)out, B_ * S_, D_, D_);
}
